// Transformer_18854906430191
// MI455X (gfx1250) — matrix operand order repaired and checked by reading, not compiled, not run on hardware
//
#include <hip/hip_runtime.h>

typedef __attribute__((ext_vector_type(16))) _Float16 v16h;
typedef __attribute__((ext_vector_type(8)))  _Float16 v8h;
typedef __attribute__((ext_vector_type(8)))  float    v8f;

#define WMMA16(a, b, c) \
  __builtin_amdgcn_wmma_f32_16x16x32_f16(false, (a), false, (b), (short)0, (c), false, false)

namespace {
constexpr int BN  = 16;    // batch
constexpr int DM  = 512;   // model dim
constexpr int KD  = 128;   // key dim
constexpr int HWN = 484;   // H*W
constexpr int TN  = 3;     // images per set
constexpr int LE  = 1452;  // encoder tokens = TN*HWN
constexpr int LEQ = 1456;  // LE padded to x16 (queries)
constexpr int LEK = 1472;  // LE padded to x64 (keys)
constexpr int LI  = 484;   // per-image tokens
constexpr int LIQ = 496;   // LI padded to x16
constexpr int LIK = 512;   // LI padded to x64
constexpr float TEMP = 30.0f;
constexpr float NORM_SCALE = 0.01104854345603981f;  // sqrt(1/(512*16))
}

__device__ __forceinline__ int laneid() { return (int)(threadIdx.x & 31u); }

// A tile 16x32 f16 (rows M = lane%16, K split per ISA layout), row-major source, ld in elems
__device__ __forceinline__ v16h load_a16(const _Float16* p, int ld) {
  int l = laneid(), r = l & 15, g = l >> 4;
  const _Float16* q = p + (size_t)r * ld + g * 8;
  v8h lo = *(const v8h*)q;
  v8h hi = *(const v8h*)(q + 16);
  v16h a;
#pragma unroll
  for (int i = 0; i < 8; ++i) { a[i] = lo[i]; a[i + 8] = hi[i]; }
  return a;
}

// B tile 32x16 f16: element (k,n) = src[n*ld + k]; lane n=l&15 loads K 8g..8g+7 and 16+8g..16+8g+7
__device__ __forceinline__ v16h load_bt(const _Float16* p, int ld) {
  int l = laneid();
  const _Float16* q = p + (size_t)(l & 15) * ld + (size_t)(l >> 4) * 8;
  v8h lo = *(const v8h*)q;
  v8h hi = *(const v8h*)(q + 16);
  v16h b;
#pragma unroll
  for (int i = 0; i < 8; ++i) { b[i] = lo[i]; b[i + 8] = hi[i]; }
  return b;
}

// ---------------- GEMM + bias + L2-normalize:  Q = l2n(X @ W^T + bias) ----------------
// X: [BN][ldbX][512] f16 (rowsPerB rows valid/padded-zero).  W16: [128][512] f16.
// Q: [BN][ldbQ][128] f16.  One wave per 16-row tile, full 128 output cols in registers.
__global__ void k_proj(const _Float16* __restrict__ X, int ldbX,
                       const _Float16* __restrict__ W16, const float* __restrict__ bias,
                       _Float16* __restrict__ Q, int ldbQ, int rowsPerB) {
  int tilesPerB = rowsPerB / 16;
  int b = blockIdx.x / tilesPerB, mt = blockIdx.x % tilesPerB;
  int l = laneid(), r = l & 15, g = l >> 4;
  const _Float16* Xb = X + ((size_t)b * ldbX + (size_t)mt * 16) * DM;
  v8f acc[8] = {};
  for (int kc = 0; kc < DM; kc += 32) {
    v16h a = load_a16(Xb + kc, DM);
#pragma unroll
    for (int n = 0; n < 8; ++n) {
      v16h w = load_bt(W16 + (size_t)(n * 16) * DM + kc, DM);
      acc[n] = WMMA16(a, w, acc[n]);
    }
  }
#pragma unroll
  for (int n = 0; n < 8; ++n) {
    float bv = bias[n * 16 + r];
#pragma unroll
    for (int j = 0; j < 8; ++j) acc[n][j] += bv;
  }
#pragma unroll
  for (int j = 0; j < 8; ++j) {
    float ss = 0.f;
#pragma unroll
    for (int n = 0; n < 8; ++n) { float v = acc[n][j]; ss += v * v; }
    ss += __shfl_xor(ss, 1, 32); ss += __shfl_xor(ss, 2, 32);
    ss += __shfl_xor(ss, 4, 32); ss += __shfl_xor(ss, 8, 32);
    float inv = 1.0f / fmaxf(sqrtf(ss), 1e-12f);
    int row = mt * 16 + j + 8 * g;
    _Float16* qr = Q + ((size_t)b * ldbQ + row) * KD;
#pragma unroll
    for (int n = 0; n < 8; ++n) qr[n * 16 + r] = (_Float16)(acc[n][j] * inv);
  }
}

// ---------------- Softmax statistics: rowsum = sum_k exp(30*q.k - 30), rowpos = sum e*pos --------
// |S| <= 30 because q,k are unit vectors, so exp(S-30) needs no running max.
__global__ void k_stats(const _Float16* __restrict__ Q, int ldbQ,
                        const _Float16* __restrict__ K, int ldbK,
                        const float* __restrict__ posw,
                        float* __restrict__ rs, float* __restrict__ rp, int rsld,
                        int Lq, int Lk, int qTiles, int nTiles) {
  int b = blockIdx.x / qTiles, qt = blockIdx.x % qTiles;
  int l = laneid(), r = l & 15, g = l >> 4;
  const _Float16* Qb = Q + ((size_t)b * ldbQ + (size_t)qt * 16) * KD;
  const _Float16* Kb = K + (size_t)b * ldbK * KD;
  const float* pwb = posw ? posw + (size_t)b * LEK : nullptr;
  v16h Aq[4];
#pragma unroll
  for (int kc = 0; kc < 4; ++kc) Aq[kc] = load_a16(Qb + kc * 32, KD);
  float sum8[8] = {}, pos8[8] = {};
  for (int kt = 0; kt < nTiles; ++kt) {
    int keybase = kt * 16;
    v8f s = {};
#pragma unroll
    for (int kc = 0; kc < 4; ++kc)
      s = WMMA16(Aq[kc], load_bt(Kb + (size_t)keybase * KD + kc * 32, KD), s);
    int key = keybase + r;
    bool ok = key < Lk;
    float pw = (pwb && ok) ? pwb[key] : 0.f;
#pragma unroll
    for (int j = 0; j < 8; ++j) {
      float e = ok ? __expf(TEMP * s[j] - TEMP) : 0.f;
      sum8[j] += e;
      pos8[j] += e * pw;
    }
  }
#pragma unroll
  for (int j = 0; j < 8; ++j) {
    float s = sum8[j], p = pos8[j];
    s += __shfl_xor(s, 1, 32); s += __shfl_xor(s, 2, 32);
    s += __shfl_xor(s, 4, 32); s += __shfl_xor(s, 8, 32);
    p += __shfl_xor(p, 1, 32); p += __shfl_xor(p, 2, 32);
    p += __shfl_xor(p, 4, 32); p += __shfl_xor(p, 8, 32);
    if (r == 0) {
      int row = qt * 16 + j + 8 * g;
      if (row < Lq) {
        rs[(size_t)b * rsld + row] = s;
        if (rp) rp[(size_t)b * rsld + row] = p;
      }
    }
  }
}

// ---------------- Cooperative AV: Out[q,:] = softmax(30*q.K) @ V ----------------
// 4 waves / workgroup, one 16-row q tile, all 512 output cols.  Per 64-key iteration:
// wave w computes the S/P tile for keys [it*64 + w*16, +16) (4 WMMAs), stages P in LDS,
// barrier, then every wave consumes the shared 16x64 P as the A operand against its own
// 128-column slice of V (16 WMMAs).  S is computed once instead of once per column group.
__global__ void k_av4(const _Float16* __restrict__ Q, int ldbQ,
                      const _Float16* __restrict__ K, int ldbK,
                      const _Float16* __restrict__ VT,
                      const float* __restrict__ rs, int rsld,
                      float* __restrict__ Out, int ldbO,
                      int Lq, int Lk, int qTiles, int nIter64) {
  int b = blockIdx.x / qTiles, qt = blockIdx.x % qTiles;
  int wid = (int)(threadIdx.x >> 5);          // wave id = column group = S-tile id
  int l = laneid(), r = l & 15, g = l >> 4;
  const _Float16* Qb = Q + ((size_t)b * ldbQ + (size_t)qt * 16) * KD;
  const _Float16* Kb = K + (size_t)b * ldbK * KD;
  const _Float16* Vb = VT + ((size_t)b * DM + (size_t)wid * 128) * ldbK;
  v16h Aq[4];
#pragma unroll
  for (int kc = 0; kc < 4; ++kc) Aq[kc] = load_a16(Qb + kc * 32, KD);
  float rinv[8];
#pragma unroll
  for (int j = 0; j < 8; ++j)
    rinv[j] = 1.0f / rs[(size_t)b * rsld + (qt * 16 + j + 8 * g)];
  v8f acc[8] = {};
  __shared__ alignas(32) _Float16 Pl[16 * 64];
  for (int it = 0; it < nIter64; ++it) {
    // --- this wave's 16-key S tile ---
    int keybase = it * 64 + wid * 16;
    if (it + 1 < nIter64)
      __builtin_prefetch((const void*)(Kb + (size_t)(it + 1) * 64 * KD +
                                       (size_t)(wid * 16 + r) * KD), 0, 1);
    v8f s = {};
#pragma unroll
    for (int kc = 0; kc < 4; ++kc)
      s = WMMA16(Aq[kc], load_bt(Kb + (size_t)keybase * KD + kc * 32, KD), s);
    int key = keybase + r;
    bool ok = key < Lk;
#pragma unroll
    for (int j = 0; j < 8; ++j) {
      float p = ok ? __expf(TEMP * s[j] - TEMP) * rinv[j] : 0.f;
      Pl[(j + 8 * g) * 64 + wid * 16 + r] = (_Float16)p;
    }
    __syncthreads();
    // --- P (16x64) @ V-slice (64x128) for this wave's column group ---
#pragma unroll
    for (int c2 = 0; c2 < 2; ++c2) {
      v16h aP;
      {
        const _Float16* pp = &Pl[(size_t)r * 64 + c2 * 32 + g * 8];
        v8h lo = *(const v8h*)pp;
        v8h hi = *(const v8h*)(pp + 16);
#pragma unroll
        for (int i = 0; i < 8; ++i) { aP[i] = lo[i]; aP[i + 8] = hi[i]; }
      }
#pragma unroll
      for (int n = 0; n < 8; ++n) {
        v16h bV = load_bt(Vb + (size_t)(n * 16) * ldbK + it * 64 + c2 * 32, ldbK);
        acc[n] = WMMA16(aP, bV, acc[n]);
      }
    }
    __syncthreads();
  }
#pragma unroll
  for (int j = 0; j < 8; ++j) {
    int row = qt * 16 + j + 8 * g;
    if (row < Lq) {
      float* orow = Out + ((size_t)b * ldbO + row) * DM + wid * 128;
#pragma unroll
      for (int n = 0; n < 8; ++n) orow[n * 16 + r] = acc[n][j];
    }
  }
}

// ---------------- elementwise / packing / norm kernels ----------------
__global__ void k_cvt_w(const float* __restrict__ sw, const float* __restrict__ cw,
                        _Float16* __restrict__ sw16, _Float16* __restrict__ cw16) {
  int i = blockIdx.x * blockDim.x + threadIdx.x;
  if (i < KD * DM) { sw16[i] = (_Float16)sw[i]; cw16[i] = (_Float16)cw[i]; }
}

__global__ void k_wts(const float* __restrict__ q, float* __restrict__ w3) {
  float a = q[0], b = q[1], c = q[2];
  float m = fmaxf(a, fmaxf(b, c));
  float ea = __expf(a - m), eb = __expf(b - m), ec = __expf(c - m);
  float s = ea + eb + ec;
  w3[0] = ea / s; w3[1] = eb / s; w3[2] = ec / s;
}

__global__ void k_pack_posw(const float* __restrict__ lab, float* __restrict__ posw) {
  size_t i = (size_t)blockIdx.x * blockDim.x + threadIdx.x;
  if (i >= (size_t)BN * LEK) return;
  int j = (int)(i % LEK), b = (int)(i / LEK);
  float v = 0.f;
  if (j < LE) v = lab[((size_t)(j / HWN) * BN + b) * HWN + (j % HWN)];
  posw[i] = v;
}

__global__ void k_pack_enc(const float* __restrict__ feat, _Float16* __restrict__ Xe,
                           _Float16* __restrict__ XeT) {
  size_t i = (size_t)blockIdx.x * blockDim.x + threadIdx.x;
  if (i >= (size_t)BN * DM * LEK) return;
  int m = (int)(i % LEK);
  size_t t1 = i / LEK;
  int d = (int)(t1 % DM), b = (int)(t1 / DM);
  float v = 0.f;
  if (m < LE) v = feat[(((size_t)(m / HWN) * BN + b) * DM + d) * HWN + (m % HWN)];
  XeT[((size_t)b * DM + d) * LEK + m] = (_Float16)v;
  if (m < LEQ) Xe[((size_t)b * LEQ + m) * DM + d] = (_Float16)v;
}

__global__ void k_pack_img(const float* __restrict__ featImg, _Float16* __restrict__ tgt16,
                           _Float16* __restrict__ tgtT) {
  size_t i = (size_t)blockIdx.x * blockDim.x + threadIdx.x;
  if (i >= (size_t)BN * DM * LIK) return;
  int m = (int)(i % LIK);
  size_t t1 = i / LIK;
  int d = (int)(t1 % DM), b = (int)(t1 / DM);
  float v = (m < LI) ? featImg[((size_t)b * DM + d) * HWN + m] : 0.f;
  tgtT[((size_t)b * DM + d) * LIK + m] = (_Float16)v;
  if (m < LIQ) tgt16[((size_t)b * LIQ + m) * DM + d] = (_Float16)v;
}

__global__ void k_addres_feat(float* __restrict__ x, int ldb,
                              const float* __restrict__ feat, int L) {
  size_t i = (size_t)blockIdx.x * blockDim.x + threadIdx.x;
  if (i >= (size_t)BN * L * DM) return;
  int d = (int)(i % DM);
  size_t t1 = i / DM;
  int m = (int)(t1 % L), b = (int)(t1 / L);
  x[((size_t)b * ldb + m) * DM + d] +=
      feat[(((size_t)(m / HWN) * BN + b) * DM + d) * HWN + (m % HWN)];
}

__global__ void k_inorm_reduce(const float* __restrict__ x, int ldb,
                               float* __restrict__ scaleOut) {
  int img = blockIdx.x / BN, b = blockIdx.x % BN;
  __shared__ float sd[256];
  float acc = 0.f;
  const int NE = HWN * DM;
  for (int i = (int)threadIdx.x; i < NE; i += 256) {
    int m = img * HWN + i / DM, d = i % DM;
    float v = x[((size_t)b * ldb + m) * DM + d];
    acc += v * v;
  }
  sd[threadIdx.x] = acc;
  __syncthreads();
  for (int s = 128; s > 0; s >>= 1) {
    if ((int)threadIdx.x < s) sd[threadIdx.x] += sd[threadIdx.x + s];
    __syncthreads();
  }
  if (threadIdx.x == 0)
    scaleOut[blockIdx.x] = NORM_SCALE * sqrtf((float)(DM * HWN) / (sd[0] + 1e-5f));
}

__global__ void k_apply_mem(const float* __restrict__ x, const float* __restrict__ scale,
                            const float* __restrict__ posw,
                            _Float16* __restrict__ mem16, _Float16* __restrict__ memPosT) {
  size_t i = (size_t)blockIdx.x * blockDim.x + threadIdx.x;
  if (i >= (size_t)BN * LEK * DM) return;
  int d = (int)(i % DM);
  size_t t1 = i / DM;
  int m = (int)(t1 % LEK), b = (int)(t1 / LEK);
  float v = 0.f;
  if (m < LE) v = x[((size_t)b * LEQ + m) * DM + d] * scale[(m / HWN) * BN + b];
  if (m < LEQ) mem16[((size_t)b * LEQ + m) * DM + d] = (_Float16)v;
  memPosT[((size_t)b * DM + d) * LEK + m] = (_Float16)(v * posw[(size_t)b * LEK + m]);
}

__global__ void k_apply_tgt(const float* __restrict__ x, const float* __restrict__ scale,
                            _Float16* __restrict__ tgt2) {
  size_t i = (size_t)blockIdx.x * blockDim.x + threadIdx.x;
  if (i >= (size_t)BN * LIQ * DM) return;
  int d = (int)(i % DM);
  size_t t1 = i / DM;
  int m = (int)(t1 % LIQ), b = (int)(t1 / LIQ);
  float v = (m < LI) ? x[((size_t)b * LIQ + m) * DM + d] * scale[b] : 0.f;
  tgt2[((size_t)b * LIQ + m) * DM + d] = (_Float16)v;
}

__global__ void k_build_valT(const _Float16* __restrict__ tgt2, const float* __restrict__ lab,
                             _Float16* __restrict__ valT) {
  size_t i = (size_t)blockIdx.x * blockDim.x + threadIdx.x;
  if (i >= (size_t)BN * DM * LEK) return;
  int j = (int)(i % LEK);
  size_t t1 = i / LEK;
  int d = (int)(t1 % DM), b = (int)(t1 / DM);
  _Float16 v = (_Float16)0.f;
  if (j < LE) {
    int p = j / HWN, k2 = j % HWN;
    v = (_Float16)(lab[((size_t)p * BN + b) * HWN + k2] *
                   (float)tgt2[((size_t)b * LIQ + k2) * DM + d]);
  }
  valT[((size_t)b * DM + d) * LEK + j] = v;
}

__global__ void k_mulmask(const _Float16* __restrict__ tgt2, const float* __restrict__ rs,
                          const float* __restrict__ rp, float* __restrict__ xb) {
  size_t i = (size_t)blockIdx.x * blockDim.x + threadIdx.x;
  if (i >= (size_t)BN * LI * DM) return;
  int d = (int)(i % DM);
  size_t t1 = i / DM;
  int m = (int)(t1 % LI), b = (int)(t1 / LI);
  float mask = rp[(size_t)b * LIQ + m] / rs[(size_t)b * LIQ + m];
  size_t xi = ((size_t)b * LIQ + m) * DM + d;
  xb[xi] = (float)tgt2[xi] * mask;
}

__global__ void k_addres16(float* __restrict__ x, const _Float16* __restrict__ res) {
  size_t i = (size_t)blockIdx.x * blockDim.x + threadIdx.x;
  if (i >= (size_t)BN * LI * DM) return;
  int d = (int)(i % DM);
  size_t t1 = i / DM;
  int m = (int)(t1 % LI), b = (int)(t1 / LI);
  size_t xi = ((size_t)b * LIQ + m) * DM + d;
  x[xi] += (float)res[xi];
}

__global__ void k_final(const float* __restrict__ xb, const float* __restrict__ xq,
                        const float* __restrict__ xv, const float* __restrict__ sb,
                        const float* __restrict__ sq, const float* __restrict__ sv,
                        const float* __restrict__ w3, float* __restrict__ out) {
  size_t i = (size_t)blockIdx.x * blockDim.x + threadIdx.x;
  if (i >= (size_t)BN * DM * HWN) return;
  int hw = (int)(i % HWN);
  size_t t1 = i / HWN;
  int d = (int)(t1 % DM), b = (int)(t1 / DM);
  size_t xi = ((size_t)b * LIQ + hw) * DM + d;
  out[i] = w3[0] * sb[b] * xb[xi] + w3[1] * sq[b] * xq[xi] + w3[2] * sv[b] * xv[xi];
}

// ---------------- host orchestration ----------------
extern "C" void kernel_launch(void* const* d_in, const int* in_sizes, int n_in,
                              void* d_out, int out_size, void* d_ws, size_t ws_size,
                              hipStream_t stream) {
  (void)in_sizes; (void)n_in; (void)out_size; (void)ws_size;
  const float* train_feat = (const float*)d_in[0];
  const float* test_feat  = (const float*)d_in[1];
  const float* train_lab  = (const float*)d_in[2];
  const float* sw  = (const float*)d_in[3];
  const float* sbv = (const float*)d_in[4];
  const float* cw  = (const float*)d_in[5];
  const float* cbv = (const float*)d_in[6];
  const float* qxm = (const float*)d_in[7];
  float* out = (float*)d_out;

  char* base = (char*)d_ws;
  size_t off = 0;
  auto alloc = [&](size_t bytes) -> void* {
    void* p = base + off;
    off += (bytes + 255) & ~(size_t)255;
    return p;
  };

  _Float16* sw16    = (_Float16*)alloc((size_t)KD * DM * 2);
  _Float16* cw16    = (_Float16*)alloc((size_t)KD * DM * 2);
  _Float16* Xe      = (_Float16*)alloc((size_t)BN * LEQ * DM * 2);
  _Float16* XeT     = (_Float16*)alloc((size_t)BN * DM * LEK * 2);
  _Float16* Qe      = (_Float16*)alloc((size_t)BN * LEK * KD * 2);
  float*    posw    = (float*)alloc((size_t)BN * LEK * 4);
  float*    encX    = (float*)alloc((size_t)BN * LEQ * DM * 4);
  float*    rs_e    = (float*)alloc((size_t)BN * LEQ * 4);
  float*    scale_e = (float*)alloc((size_t)TN * BN * 4);
  _Float16* mem16   = (_Float16*)alloc((size_t)BN * LEQ * DM * 2);
  _Float16* memPosT = (_Float16*)alloc((size_t)BN * DM * LEK * 2);
  _Float16* Kc      = (_Float16*)alloc((size_t)BN * LEK * KD * 2);
  float*    w3      = (float*)alloc(16);
  _Float16* tgt16   = (_Float16*)alloc((size_t)BN * LIQ * DM * 2);
  _Float16* tgtT16  = (_Float16*)alloc((size_t)BN * DM * LIK * 2);
  _Float16* Qi      = (_Float16*)alloc((size_t)BN * LIK * KD * 2);
  _Float16* Qc      = (_Float16*)alloc((size_t)BN * LIK * KD * 2);
  _Float16* tgt2    = (_Float16*)alloc((size_t)BN * LIQ * DM * 2);
  _Float16* valT    = (_Float16*)alloc((size_t)BN * DM * LEK * 2);
  float*    xb      = (float*)alloc((size_t)BN * LIQ * DM * 4);
  float*    xq      = (float*)alloc((size_t)BN * LIQ * DM * 4);
  float*    xv      = (float*)alloc((size_t)BN * LIQ * DM * 4);
  float*    rs_i    = (float*)alloc((size_t)BN * LIQ * 4);
  float*    rs_c    = (float*)alloc((size_t)BN * LIQ * 4);
  float*    rp_c    = (float*)alloc((size_t)BN * LIQ * 4);
  float*    scale_i = (float*)alloc((size_t)BN * 4);
  float*    s_b     = (float*)alloc((size_t)BN * 4);
  float*    s_q     = (float*)alloc((size_t)BN * 4);
  float*    s_v     = (float*)alloc((size_t)BN * 4);

  auto ew = [](size_t n) { return dim3((unsigned)((n + 255) / 256)); };
  const dim3 TB(256), W32(32), W128(128);

  // ---- shared prep ----
  k_cvt_w<<<ew((size_t)KD * DM), TB, 0, stream>>>(sw, cw, sw16, cw16);
  k_wts<<<dim3(1), dim3(1), 0, stream>>>(qxm, w3);
  k_pack_posw<<<ew((size_t)BN * LEK), TB, 0, stream>>>(train_lab, posw);

  // ---- encoder: memory = inorm(src + selfattn(src)) ----
  k_pack_enc<<<ew((size_t)BN * DM * LEK), TB, 0, stream>>>(train_feat, Xe, XeT);
  k_proj<<<dim3(BN * (LEQ / 16)), W32, 0, stream>>>(Xe, LEQ, sw16, sbv, Qe, LEK, LEQ);
  k_stats<<<dim3(BN * (LEQ / 16)), W32, 0, stream>>>(Qe, LEK, Qe, LEK, nullptr,
                                                     rs_e, nullptr, LEQ, LE, LE,
                                                     LEQ / 16, LEK / 16);
  k_av4<<<dim3(BN * (LEQ / 16)), W128, 0, stream>>>(Qe, LEK, Qe, LEK, XeT, rs_e, LEQ,
                                                    encX, LEQ, LE, LE, LEQ / 16, LEK / 64);
  k_addres_feat<<<ew((size_t)BN * LE * DM), TB, 0, stream>>>(encX, LEQ, train_feat, LE);
  k_inorm_reduce<<<dim3(TN * BN), TB, 0, stream>>>(encX, LEQ, scale_e);
  k_apply_mem<<<ew((size_t)BN * LEK * DM), TB, 0, stream>>>(encX, scale_e, posw, mem16, memPosT);
  // cross keys from memory (shared across all 6 decodes)
  k_proj<<<dim3(BN * (LEQ / 16)), W32, 0, stream>>>(mem16, LEQ, cw16, cbv, Kc, LEK, LEQ);

  // ---- decode each image (3 enc + 3 dec) ----
  for (int img = 0; img < 6; ++img) {
    const float* featImg =
        (img < 3 ? train_feat : test_feat) + (size_t)(img % 3) * BN * DM * HWN;
    float* outImg = out + (size_t)img * BN * DM * HWN;

    k_pack_img<<<ew((size_t)BN * DM * LIK), TB, 0, stream>>>(featImg, tgt16, tgtT16);
    // self-attention + residual + inorm -> tgt2
    k_proj<<<dim3(BN * (LIQ / 16)), W32, 0, stream>>>(tgt16, LIQ, sw16, sbv, Qi, LIK, LIQ);
    k_stats<<<dim3(BN * (LIQ / 16)), W32, 0, stream>>>(Qi, LIK, Qi, LIK, nullptr,
                                                       rs_i, nullptr, LIQ, LI, LI,
                                                       LIQ / 16, LIK / 16);
    k_av4<<<dim3(BN * (LIQ / 16)), W128, 0, stream>>>(Qi, LIK, Qi, LIK, tgtT16, rs_i, LIQ,
                                                      xb, LIQ, LI, LI, LIQ / 16, LIK / 64);
    k_addres_feat<<<ew((size_t)BN * LI * DM), TB, 0, stream>>>(xb, LIQ, featImg, LI);
    k_inorm_reduce<<<dim3(BN), TB, 0, stream>>>(xb, LIQ, scale_i);
    k_apply_tgt<<<ew((size_t)BN * LIQ * DM), TB, 0, stream>>>(xb, scale_i, tgt2);
    // cross-attention probabilities (shared by mask / t_q / t_v)
    k_proj<<<dim3(BN * (LIQ / 16)), W32, 0, stream>>>(tgt2, LIQ, cw16, cbv, Qc, LIK, LIQ);
    k_stats<<<dim3(BN * (LIQ / 16)), W32, 0, stream>>>(Qc, LIK, Kc, LEK, posw,
                                                       rs_c, rp_c, LIQ, LI, LE,
                                                       LIQ / 16, LEK / 16);
    k_build_valT<<<ew((size_t)BN * DM * LEK), TB, 0, stream>>>(tgt2, train_lab, valT);
    k_av4<<<dim3(BN * (LIQ / 16)), W128, 0, stream>>>(Qc, LIK, Kc, LEK, valT, rs_c, LIQ,
                                                      xq, LIQ, LI, LE, LIQ / 16, LEK / 64);
    k_av4<<<dim3(BN * (LIQ / 16)), W128, 0, stream>>>(Qc, LIK, Kc, LEK, memPosT, rs_c, LIQ,
                                                      xv, LIQ, LI, LE, LIQ / 16, LEK / 64);
    // three branches + their instance norms
    k_mulmask<<<ew((size_t)BN * LI * DM), TB, 0, stream>>>(tgt2, rs_c, rp_c, xb);
    k_addres16<<<ew((size_t)BN * LI * DM), TB, 0, stream>>>(xq, tgt2);
    k_addres16<<<ew((size_t)BN * LI * DM), TB, 0, stream>>>(xv, tgt2);
    k_inorm_reduce<<<dim3(BN), TB, 0, stream>>>(xb, LIQ, s_b);
    k_inorm_reduce<<<dim3(BN), TB, 0, stream>>>(xq, LIQ, s_q);
    k_inorm_reduce<<<dim3(BN), TB, 0, stream>>>(xv, LIQ, s_v);
    k_final<<<ew((size_t)BN * DM * HWN), TB, 0, stream>>>(xb, xq, xv, s_b, s_q, s_v, w3, outImg);
  }
}
